// InfoNCEHardNegativeLossMemOpt_83760452206739
// MI455X (gfx1250) — matrix operand order repaired and checked by reading, not compiled, not run on hardware
//
#include <hip/hip_runtime.h>
#include <hip/hip_bf16.h>
#include <math.h>

typedef __attribute__((ext_vector_type(16))) __bf16 v16bf;
typedef __attribute__((ext_vector_type(8)))  float  v8f;

#define WAVE   32
#define DIMD   256
#define NCHUNK 8

// ---------- helpers ----------
static __device__ __forceinline__ unsigned short f32_to_bf16_rne(float f) {
  unsigned int u = __float_as_uint(f);
  u += 0x7FFFu + ((u >> 16) & 1u);
  return (unsigned short)(u >> 16);
}
static __device__ __forceinline__ float bf16_bits_to_f32(unsigned short h) {
  return __uint_as_float(((unsigned int)h) << 16);
}
// load 16 bf16 as two 16B vectors: 8 at p, 8 at p+off2 (elements)
static __device__ __forceinline__ v16bf load2x8(const unsigned short* p, int off2) {
  union { v16bf v; uint4 u[2]; } t;
  t.u[0] = *reinterpret_cast<const uint4*>(p);
  t.u[1] = *reinterpret_cast<const uint4*>(p + off2);
  return t.v;
}

// ---------- kernel 1: L2-normalize + bf16 hi/lo split ----------
__global__ __launch_bounds__(256)
void normalize_split_kernel(const float* __restrict__ anc,
                            const float* __restrict__ pos,
                            const float* __restrict__ neg,
                            unsigned short* __restrict__ ahi,
                            unsigned short* __restrict__ alo,
                            unsigned short* __restrict__ thi,
                            unsigned short* __restrict__ tlo,
                            int B) {
  const int lane = threadIdx.x & (WAVE - 1);
  const int row  = blockIdx.x * (blockDim.x / WAVE) + (threadIdx.x >> 5);
  if (row >= 3 * B) return;

  const float* src;
  unsigned short *dhi, *dlo;
  int srow, orow;
  if (row < B)            { src = anc; srow = row;         orow = row;     dhi = ahi; dlo = alo; }
  else if (row < 2 * B)   { src = pos; srow = row - B;     orow = row - B; dhi = thi; dlo = tlo; }
  else                    { src = neg; srow = row - 2 * B; orow = row - B; dhi = thi; dlo = tlo; }

  const float* rp = src + (size_t)srow * DIMD;
  float v[8];
  float ss = 0.f;
#pragma unroll
  for (int k = 0; k < 8; ++k) {
    float x = rp[lane + 32 * k];
    v[k] = x;
    ss += x * x;
  }
#pragma unroll
  for (int off = 16; off >= 1; off >>= 1) ss += __shfl_xor(ss, off, WAVE);
  const float scale = 1.0f / fmaxf(sqrtf(ss), 1e-12f);

  unsigned short* hp = dhi + (size_t)orow * DIMD;
  unsigned short* lp = dlo + (size_t)orow * DIMD;
#pragma unroll
  for (int k = 0; k < 8; ++k) {
    float xs = v[k] * scale;
    unsigned short h = f32_to_bf16_rne(xs);
    float r = xs - bf16_bits_to_f32(h);
    hp[lane + 32 * k] = h;
    lp[lane + 32 * k] = f32_to_bf16_rne(r);
  }
}

// ---------- kernel 2: fused bf16x3 WMMA GEMM + online logsumexp ----------
// Software-pipelined: B-panel loads for chunk k+1 are issued before the WMMAs
// consuming chunk k, so the scheduler can use partial s_wait_loadcnt values.
__global__ __launch_bounds__(256)
void lse_gemm_kernel(const unsigned short* __restrict__ ahi,
                     const unsigned short* __restrict__ alo,
                     const unsigned short* __restrict__ thi,
                     const unsigned short* __restrict__ tlo,
                     float* __restrict__ diag,
                     float* __restrict__ pM,
                     float* __restrict__ pS,
                     int B) {
  const int lane    = threadIdx.x & 31;
  const int wave    = (int)((blockIdx.x * blockDim.x + threadIdx.x) >> 5);
  const int rowTile = wave >> 3;            // B/16 row tiles
  const int chunk   = wave & (NCHUNK - 1);
  const int rowBase = rowTile * 16;
  const int colsPerChunk = (2 * B) / NCHUNK;
  const int colStart     = chunk * colsPerChunk;
  const int nTiles       = colsPerChunk / 16;

  // A operand: lane&15 = M, 16 contiguous K per half-wave
  const int mrow  = rowBase + (lane & 15);
  const int koffA = (lane >> 4) * 16;
  v16bf aH[8], aL[8];
#pragma unroll
  for (int kc = 0; kc < 8; ++kc) {
    aH[kc] = load2x8(ahi + (size_t)mrow * DIMD + kc * 32 + koffA, 8);
    aL[kc] = load2x8(alo + (size_t)mrow * DIMD + kc * 32 + koffA, 8);
  }

  // B operand: lane&15 = N, 16 contiguous K per half-wave
  const int koffB = (lane >> 4) * 16;
  const unsigned short* bhBase = thi + (size_t)(colStart + (lane & 15)) * DIMD + koffB;
  const unsigned short* blBase = tlo + (size_t)(colStart + (lane & 15)) * DIMD + koffB;
  const size_t tileStep = (size_t)16 * DIMD;   // element step per column tile

  float m[8], s[8];
#pragma unroll
  for (int p = 0; p < 8; ++p) { m[p] = -1e30f; s[p] = 0.f; }

  const float K2 = 28.853900817779268f;  // log2(e) / 0.05
  const int diagP = (lane < 8) ? lane : ((lane >= 24) ? (lane - 24) : -1);
  float diagv = 0.f;
  int haveDiag = 0;

  // prologue: chunk 0 of tile 0
  v16bf bH = load2x8(bhBase, 8);
  v16bf bL = load2x8(blBase, 8);

  for (int ct = 0; ct < nTiles; ++ct) {
    const int    colBase = colStart + ct * 16;
    const size_t curOff  = (size_t)ct * tileStep;
    const size_t nxtOff  = (size_t)((ct + 1 < nTiles) ? (ct + 1) : ct) * tileStep;

    v8f c  = {};
    v8f c2 = {};
#pragma unroll
    for (int kc = 0; kc < 8; ++kc) {
      // issue loads for the NEXT chunk (or next tile's chunk 0) first
      const size_t noff = (kc < 7) ? (curOff + (size_t)(kc + 1) * 32) : nxtOff;
      v16bf nH = load2x8(bhBase + noff, 8);
      v16bf nL = load2x8(blBase + noff, 8);
      c  = __builtin_amdgcn_wmma_f32_16x16x32_bf16(false, aH[kc], false, bH, (short)0, c,  false, false);
      c  = __builtin_amdgcn_wmma_f32_16x16x32_bf16(false, aL[kc], false, bH, (short)0, c,  false, false);
      c2 = __builtin_amdgcn_wmma_f32_16x16x32_bf16(false, aH[kc], false, bL, (short)0, c2, false, false);
      bH = nH;
      bL = nL;
    }
    if (ct + 2 < nTiles) {  // warm L2 two tiles ahead
      __builtin_prefetch(bhBase + curOff + 2 * tileStep, 0, 1);
      __builtin_prefetch(blBase + curOff + 2 * tileStep, 0, 1);
    }

    const int isDiag = (colBase == rowBase);
#pragma unroll
    for (int p = 0; p < 8; ++p) {
      float dot = c[p] + c2[p];
      float x   = dot * K2;                 // base-2 logit
      float nm  = fmaxf(m[p], x);
      s[p] = s[p] * exp2f(m[p] - nm) + exp2f(x - nm);
      m[p] = nm;
      if (isDiag && p == diagP) { diagv = dot; haveDiag = 1; }
    }
  }

  // butterfly merge of (max,sum) across each 16-lane half (N direction)
#pragma unroll
  for (int p = 0; p < 8; ++p) {
    float mm = m[p], ss = s[p];
    for (int off = 1; off < 16; off <<= 1) {
      float mo = __shfl_xor(mm, off, WAVE);
      float so = __shfl_xor(ss, off, WAVE);
      float nm = fmaxf(mm, mo);
      ss = ss * exp2f(mm - nm) + so * exp2f(mo - nm);
      mm = nm;
    }
    m[p] = mm; s[p] = ss;
  }

  if ((lane & 15) == 0) {
    const int half = lane >> 4;
#pragma unroll
    for (int p = 0; p < 8; ++p) {
      const int row = rowBase + half * 8 + p;
      pM[row * NCHUNK + chunk] = m[p];
      pS[row * NCHUNK + chunk] = s[p];
    }
  }
  if (haveDiag) {
    const int row = rowBase + ((lane < 8) ? lane : (lane - 16));
    diag[row] = diagv;
  }
}

// ---------- kernel 3: deterministic final reduction ----------
__global__ __launch_bounds__(256)
void finalize_kernel(const float* __restrict__ diag,
                     const float* __restrict__ pM,
                     const float* __restrict__ pS,
                     float* __restrict__ out, int B) {
  __shared__ float red[256];
  float acc = 0.f;
  for (int r = (int)threadIdx.x; r < B; r += 256) {
    float M = -1e30f, S = 0.f;
#pragma unroll
    for (int c = 0; c < NCHUNK; ++c) {
      float mc = pM[r * NCHUNK + c];
      float sc = pS[r * NCHUNK + c];
      float nm = fmaxf(M, mc);
      S = S * exp2f(M - nm) + sc * exp2f(mc - nm);
      M = nm;
    }
    float lse_ln = (M + log2f(S)) * 0.69314718055994531f;
    acc += lse_ln - diag[r] * 20.0f;  // 1/TEMPERATURE
  }
  red[threadIdx.x] = acc;
  __syncthreads();
  for (int off = 128; off >= 1; off >>= 1) {
    if ((int)threadIdx.x < off) red[threadIdx.x] += red[threadIdx.x + off];
    __syncthreads();
  }
  if (threadIdx.x == 0) out[0] = red[0] / (float)B;
}

// ---------- launcher ----------
extern "C" void kernel_launch(void* const* d_in, const int* in_sizes, int n_in,
                              void* d_out, int out_size, void* d_ws, size_t ws_size,
                              hipStream_t stream) {
  (void)n_in; (void)out_size; (void)ws_size;
  const float* anc = (const float*)d_in[0];
  const float* pos = (const float*)d_in[1];
  const float* neg = (const float*)d_in[2];
  const int B = in_sizes[0] / DIMD;  // 8192

  char* ws = (char*)d_ws;
  size_t off = 0;
  auto alloc = [&](size_t bytes) -> void* {
    off = (off + 255) & ~(size_t)255;
    void* p = ws + off;
    off += bytes;
    return p;
  };
  unsigned short* ahi = (unsigned short*)alloc((size_t)B * DIMD * 2);
  unsigned short* alo = (unsigned short*)alloc((size_t)B * DIMD * 2);
  unsigned short* thi = (unsigned short*)alloc((size_t)2 * B * DIMD * 2);
  unsigned short* tlo = (unsigned short*)alloc((size_t)2 * B * DIMD * 2);
  float* diag = (float*)alloc((size_t)B * 4);
  float* pM   = (float*)alloc((size_t)B * NCHUNK * 4);
  float* pS   = (float*)alloc((size_t)B * NCHUNK * 4);

  // 1) normalize + split: 3B rows, 8 rows (waves) per 256-thread block
  const int rowsTotal = 3 * B;
  normalize_split_kernel<<<(rowsTotal + 7) / 8, 256, 0, stream>>>(
      anc, pos, neg, ahi, alo, thi, tlo, B);

  // 2) fused WMMA GEMM + online LSE: (B/16) row tiles x NCHUNK chunks, 8 waves/block
  const int totalWaves = (B / 16) * NCHUNK;  // 4096
  lse_gemm_kernel<<<totalWaves / 8, 256, 0, stream>>>(
      ahi, alo, thi, tlo, diag, pM, pS, B);

  // 3) deterministic scalar reduction
  finalize_kernel<<<1, 256, 0, stream>>>(diag, pM, pS, (float*)d_out, B);
}
